// SoftMoEDecoderLayer_48395691491542
// MI455X (gfx1250) — matrix operand order repaired and checked by reading, not compiled, not run on hardware
//
#include <hip/hip_runtime.h>

// ---------------------------------------------------------------------------
// SoftMoE decoder layer for MI455X (gfx1250, CDNA5, wave32).
// All GEMMs run on v_wmma_f32_16x16x32_bf16. Operands converted to bf16 once;
// GEMM K-loop is double-buffered in LDS with GLOBAL_LOAD_ASYNC_TO_LDS_B128
// (ASYNCcnt-tracked), overlapping the tile t+1 copy with the WMMAs of tile t.
// ---------------------------------------------------------------------------

#if defined(__AMDGCN__) && __has_builtin(__builtin_amdgcn_global_load_async_to_lds_b128) && __has_builtin(__builtin_amdgcn_s_wait_asynccnt)
#define USE_ASYNC_LDS 1
#else
#define USE_ASYNC_LDS 0
#endif

typedef __bf16 bf16;
typedef __attribute__((ext_vector_type(16))) __bf16 v16bf;
typedef __attribute__((ext_vector_type(8)))  __bf16 v8bf;
typedef __attribute__((ext_vector_type(8)))  float   v8f;
typedef int v4i_raw __attribute__((vector_size(16)));   // matches builtin param

namespace cfg {
constexpr int B  = 4;
constexpr int M  = 1024;
constexpr int S  = 1024;
constexpr int D  = 1024;
constexpr int H  = 16;
constexpr int FF = 4096;
constexpr int NE = 128;
constexpr int HD = D / H;          // 64
constexpr float EPS = 1e-5f;
}

template <typename T> struct is_bf16_t { static constexpr bool value = false; };
template <> struct is_bf16_t<bf16>     { static constexpr bool value = true;  };

static __device__ __forceinline__ unsigned short bfbits(float f) {
  union { float f; unsigned u; } a; a.f = f;
  unsigned r = a.u + 0x7FFFu + ((a.u >> 16) & 1u);   // round-to-nearest-even
  return (unsigned short)(r >> 16);
}
static __device__ __forceinline__ bf16 f2bf(float f) {
  union { unsigned short s; bf16 b; } c; c.s = bfbits(f);
  return c.b;
}
static __device__ __forceinline__ unsigned pack2(float x, float y) {
  return (unsigned)bfbits(x) | ((unsigned)bfbits(y) << 16);
}

static __device__ __forceinline__ float wave_max(float v) {
  for (int o = 16; o; o >>= 1) v = fmaxf(v, __shfl_xor(v, o, 32));
  return v;
}
static __device__ __forceinline__ float wave_sum(float v) {
  for (int o = 16; o; o >>= 1) v += __shfl_xor(v, o, 32);
  return v;
}

// 16-byte global -> LDS copy: async (ASYNCcnt-tracked) or sync fallback.
static __device__ __forceinline__ void copy16_g2l(const bf16* g, bf16* l) {
#if USE_ASYNC_LDS
  typedef __attribute__((address_space(1))) v4i_raw gv4;   // global
  typedef __attribute__((address_space(3))) v4i_raw lv4;   // LDS
  __builtin_amdgcn_global_load_async_to_lds_b128((gv4*)g, (lv4*)l, 0, 0);
#else
  *(uint4*)l = *(const uint4*)g;
#endif
}
static __device__ __forceinline__ void wait_async_copies() {
#if USE_ASYNC_LDS
  __builtin_amdgcn_s_wait_asynccnt(0);
#endif
}

// ---------------------------------------------------------------------------
// Main WMMA GEMM: C = A(bf16) @ B(bf16) (+bias f32) [+relu], CT in {float,bf16}
//   BT==true : B supplied as (N,K) row-major (computes A @ B^T)
//   BT==false: B supplied as (K,N) row-major
// 256 threads = 8 waves; block tile 128x128x64; wave tile 64x32 (4x2 WMMA).
// LDS double-buffered; async copies for tile t+1 overlap the WMMAs of tile t.
// Kdim must be a multiple of 64 (true for all call sites here).
// ---------------------------------------------------------------------------
template <bool BT, bool RELU, typename CT>
__global__ __launch_bounds__(256) void gemm_bf16_kernel(
    const bf16* __restrict__ A, int lda, long long sAb,
    const bf16* __restrict__ Bsrc, int ldb, long long sBb,
    const float* __restrict__ bias,
    CT* __restrict__ C, int ldc, long long sCb,
    int Mdim, int Ndim, int Kdim)
{
  constexpr int BK = 64;
  __shared__ bf16 lsA[2][128 * BK];   // [m][k]  2 x 16 KB
  __shared__ bf16 lsB[2][BK * 128];   // [k][n]  2 x 16 KB

  const int z = blockIdx.z;
  const bf16* Ap = A + (long long)z * sAb;
  const bf16* Bp = Bsrc + (long long)z * sBb;
  CT* Cp = C + (long long)z * sCb;

  const int tileM = blockIdx.y * 128;
  const int tileN = blockIdx.x * 128;
  const int tid = threadIdx.x, lane = tid & 31, wave = tid >> 5;
  const int waveM = wave >> 2, waveN = wave & 3;

  v8f acc[4][2] = {};

  auto issueA = [&](int k0, int buf) {
    for (int p = 0; p < 4; ++p) {
      int id = tid + p * 256;
      int row = id >> 3, ch = id & 7;
      int gm = tileM + row; if (gm >= Mdim) gm = Mdim - 1;
      const bf16* g = Ap + (long long)gm * lda + k0 + ch * 8;
      copy16_g2l(g, &lsA[buf][row * BK + ch * 8]);
      __builtin_prefetch(g + 2 * BK, 0, 0);
    }
  };
  auto issueB = [&](int k0, int buf) {
    if (BT) {
      // (N,K) weights: load along K, scatter-transpose into [k][n] (sync path;
      // only attention K^T uses this and it is a single K-tile of 64).
      for (int p = 0; p < 4; ++p) {
        int id = tid + p * 256;
        int n = id >> 3, ch = id & 7;
        int gn = tileN + n; if (gn >= Ndim) gn = Ndim - 1;
        union { uint4 u; bf16 h[8]; } t;
        t.u = *(const uint4*)(Bp + (long long)gn * ldb + k0 + ch * 8);
        int k = ch * 8;
        for (int j = 0; j < 8; ++j) lsB[buf][(k + j) * 128 + n] = t.h[j];
      }
    } else {
      for (int p = 0; p < 4; ++p) {
        int id = tid + p * 256;
        int row = id >> 4, ch = id & 15;
        int gn = tileN + ch * 8; if (gn >= Ndim) gn = 0;
        const bf16* g = Bp + (long long)(k0 + row) * ldb + gn;
        copy16_g2l(g, &lsB[buf][row * 128 + ch * 8]);
        __builtin_prefetch(g + (long long)2 * BK * ldb, 0, 0);
      }
    }
  };

  const int nTiles = Kdim / BK;
  issueA(0, 0);
  issueB(0, 0);
  for (int t = 0; t < nTiles; ++t) {
    const int cur = t & 1;
    wait_async_copies();       // tile t fully landed (this wave's share)
    __syncthreads();           // all waves' shares landed
    if (t + 1 < nTiles) {      // overlap tile t+1 copy with tile t WMMAs
      issueA((t + 1) * BK, cur ^ 1);
      issueB((t + 1) * BK, cur ^ 1);
    }

    for (int kk = 0; kk < BK; kk += 32) {
      v16bf afr[4];
      const int kg = kk + (lane >> 4) * 8;
      for (int i = 0; i < 4; ++i) {
        int m = waveM * 64 + i * 16 + (lane & 15);
        v8bf lo = *(const v8bf*)&lsA[cur][m * BK + kg];
        v8bf hi = *(const v8bf*)&lsA[cur][m * BK + kg + 16];
        v16bf a;
        for (int e = 0; e < 8; ++e) { a[e] = lo[e]; a[e + 8] = hi[e]; }
        afr[i] = a;
      }
      for (int j = 0; j < 2; ++j) {
        int nb = waveN * 32 + j * 16;
        v16bf bfr;
        for (int e = 0; e < 8; ++e) {
          bfr[e]     = lsB[cur][(kg + e) * 128 + nb + (lane & 15)];
          bfr[e + 8] = lsB[cur][(kg + 16 + e) * 128 + nb + (lane & 15)];
        }
        for (int i = 0; i < 4; ++i) {
          acc[i][j] = __builtin_amdgcn_wmma_f32_16x16x32_bf16(
              false, afr[i], false, bfr, (short)0, acc[i][j], false, false);
        }
      }
    }
    __syncthreads();           // done reading buf cur before t+2 overwrites it
  }

  for (int j = 0; j < 2; ++j) {
    int col = tileN + waveN * 32 + j * 16 + (lane & 15);
    if (col >= Ndim) continue;
    float bv = bias ? bias[col] : 0.f;
    for (int i = 0; i < 4; ++i) {
      int rbase = tileM + waveM * 64 + i * 16 + (lane >> 4) * 8;
      for (int r = 0; r < 8; ++r) {
        int row = rbase + r;
        if (row < Mdim) {
          float v = acc[i][j][r] + bv;
          if (RELU) v = fmaxf(v, 0.f);
          if constexpr (is_bf16_t<CT>::value)
            Cp[(long long)row * ldc + col] = f2bf(v);
          else
            Cp[(long long)row * ldc + col] = v;
        }
      }
    }
  }
}

// ---------------------------------------------------------------------------
// Small-M expert GEMM (per expert n): y[b,n] = slots[b,n] @ exp_w[n] + exp_b[n]
// BM=16 (M=4 valid rows), BN=128, BK=64; 8 waves, one 16x16 WMMA tile each.
// A is bf16; B (exp_w, reuse=1) converted f32->bf16 in-kernel.
// ---------------------------------------------------------------------------
__global__ __launch_bounds__(256) void gemm_expert_kernel(
    const bf16* __restrict__ A,      // slots_bf (B, NE, FF)
    const float* __restrict__ Bsrc,  // exp_w (NE, FF, D)
    const float* __restrict__ bias,  // exp_b (NE, D)
    bf16* __restrict__ C,            // ybuf (B, NE, D)
    int Mdim, int Ndim, int Kdim)
{
  using namespace cfg;
  constexpr int BK = 64;
  __shared__ bf16 lsA[16 * BK];      // 2 KB
  __shared__ bf16 lsB[BK * 128];     // 16 KB

  const int n = blockIdx.z;
  const bf16*  Ap    = A + (long long)n * FF;        // row stride NE*FF
  const float* Bp    = Bsrc + (long long)n * FF * D;
  const float* biasp = bias + (long long)n * D;
  bf16*        Cp    = C + (long long)n * D;         // row stride NE*D
  const int lda = NE * FF, ldb = D, ldc = NE * D;

  const int tileN = blockIdx.x * 128;
  const int tid = threadIdx.x, lane = tid & 31, wave = tid >> 5;

  v8f acc = {};
  for (int k0 = 0; k0 < Kdim; k0 += BK) {
    if (tid < 128) {
      int row = tid >> 3, ch = tid & 7;
      int gm = (row < Mdim) ? row : 0;
      *(uint4*)&lsA[row * BK + ch * 8] =
          *(const uint4*)(Ap + (long long)gm * lda + k0 + ch * 8);
    }
    for (int p = 0; p < 8; ++p) {
      int id = tid + p * 256;
      int row = id >> 5, ch = id & 31;
      int gn = tileN + ch * 4; if (gn >= Ndim) gn = 0;
      float4 v = *(const float4*)(Bp + (long long)(k0 + row) * ldb + gn);
      *(uint2*)&lsB[row * 128 + ch * 4] = make_uint2(pack2(v.x, v.y), pack2(v.z, v.w));
    }
    __syncthreads();
    for (int kk = 0; kk < BK; kk += 32) {
      const int kg = kk + (lane >> 4) * 8;
      int m = lane & 15;
      v8bf lo = *(const v8bf*)&lsA[m * BK + kg];
      v8bf hi = *(const v8bf*)&lsA[m * BK + kg + 16];
      v16bf a;
      for (int e = 0; e < 8; ++e) { a[e] = lo[e]; a[e + 8] = hi[e]; }
      v16bf b;
      for (int e = 0; e < 8; ++e) {
        b[e]     = lsB[(kg + e) * 128 + wave * 16 + m];
        b[e + 8] = lsB[(kg + 16 + e) * 128 + wave * 16 + m];
      }
      acc = __builtin_amdgcn_wmma_f32_16x16x32_bf16(
          false, a, false, b, (short)0, acc, false, false);
    }
    __syncthreads();
  }
  int col = tileN + wave * 16 + (lane & 15);
  if (col < Ndim) {
    float bv = biasp[col];
    for (int r = 0; r < 8; ++r) {
      int row = r + (lane >> 4) * 8;
      if (row < Mdim) Cp[(long long)row * ldc + col] = f2bf(acc[r] + bv);
    }
  }
}

// ---------------------------------------------------------------------------
// f32 -> bf16 bulk convert (vectorized, grid-stride over float4 chunks)
// ---------------------------------------------------------------------------
__global__ __launch_bounds__(256) void convert_bf16_kernel(
    const float* __restrict__ src, bf16* __restrict__ dst, long long n4)
{
  long long i = (long long)blockIdx.x * 256 + threadIdx.x;
  const long long stride = (long long)gridDim.x * 256;
  for (; i < n4; i += stride) {
    float4 v = *(const float4*)(src + i * 4);
    *(uint2*)(dst + i * 4) = make_uint2(pack2(v.x, v.y), pack2(v.z, v.w));
  }
}

// ---------------------------------------------------------------------------
// (N,K) f32 -> (K,N) bf16 transposed convert (32x32 LDS tile)
// ---------------------------------------------------------------------------
__global__ __launch_bounds__(256) void transpose_convert_kernel(
    const float* __restrict__ src, bf16* __restrict__ dst, int N, int K)
{
  __shared__ float t[32][33];
  const int tx = threadIdx.x & 31;
  const int ty = threadIdx.x >> 5;
  const int n0 = blockIdx.y * 32, k0 = blockIdx.x * 32;
  for (int j = 0; j < 4; ++j) {
    int n = n0 + ty + j * 8, k = k0 + tx;
    float v = 0.f;
    if (n < N && k < K) v = src[(long long)n * K + k];
    t[ty + j * 8][tx] = v;
  }
  __syncthreads();
  for (int j = 0; j < 4; ++j) {
    int k = k0 + ty + j * 8, n = n0 + tx;
    if (k < K && n < N) dst[(long long)k * N + n] = f2bf(t[tx][ty + j * 8]);
  }
}

// ---------------------------------------------------------------------------
// Row softmax (f32 in, bf16 out): out[row] = softmax(in[row] * scale)
// ---------------------------------------------------------------------------
__global__ __launch_bounds__(256) void softmax_rows_kernel(
    const float* __restrict__ in, bf16* __restrict__ out, int L, float scale)
{
  __shared__ float sh[8];
  const long long row = blockIdx.x;
  const float* ip = in + row * L;
  bf16* op = out + row * L;
  const int tid = threadIdx.x, lane = tid & 31, wid = tid >> 5;

  float mx = -1e30f;
  for (int i = tid; i < L; i += 256) mx = fmaxf(mx, ip[i] * scale);
  mx = wave_max(mx);
  if (lane == 0) sh[wid] = mx;
  __syncthreads();
  float m = sh[0];
  for (int i = 1; i < 8; ++i) m = fmaxf(m, sh[i]);
  __syncthreads();

  float s = 0.f;
  for (int i = tid; i < L; i += 256) s += __expf(ip[i] * scale - m);
  s = wave_sum(s);
  if (lane == 0) sh[wid] = s;
  __syncthreads();
  float tot = 0.f;
  for (int i = 0; i < 8; ++i) tot += sh[i];
  const float inv = 1.f / tot;
  for (int i = tid; i < L; i += 256) op[i] = f2bf(__expf(ip[i] * scale - m) * inv);
}

// ---------------------------------------------------------------------------
// Dispatch softmax over token axis per (b, expert), transposed bf16 output
// dispT[b][n][m] (GEMM A-operand for the slots GEMM).
// ---------------------------------------------------------------------------
__global__ __launch_bounds__(256) void dispatch_softmax_kernel(
    const float* __restrict__ logits, bf16* __restrict__ dispT, int Mrows)
{
  __shared__ float sh[8];
  const int b = blockIdx.x / cfg::NE;
  const int n = blockIdx.x % cfg::NE;
  const float* ip = logits + (long long)b * Mrows * cfg::NE + n;   // stride NE
  bf16* op = dispT + ((long long)b * cfg::NE + n) * Mrows;
  const int tid = threadIdx.x, lane = tid & 31, wid = tid >> 5;

  float mx = -1e30f;
  for (int i = tid; i < Mrows; i += 256)
    mx = fmaxf(mx, ip[(long long)i * cfg::NE]);
  mx = wave_max(mx);
  if (lane == 0) sh[wid] = mx;
  __syncthreads();
  float m = sh[0];
  for (int i = 1; i < 8; ++i) m = fmaxf(m, sh[i]);
  __syncthreads();

  float s = 0.f;
  for (int i = tid; i < Mrows; i += 256)
    s += __expf(ip[(long long)i * cfg::NE] - m);
  s = wave_sum(s);
  if (lane == 0) sh[wid] = s;
  __syncthreads();
  float tot = 0.f;
  for (int i = 0; i < 8; ++i) tot += sh[i];
  const float inv = 1.f / tot;
  for (int i = tid; i < Mrows; i += 256)
    op[i] = f2bf(__expf(ip[(long long)i * cfg::NE] - m) * inv);
}

// ---------------------------------------------------------------------------
// out = LayerNorm(resid + y) * g + beta ; writes f32 (+ optional bf16 copy)
// ---------------------------------------------------------------------------
__global__ __launch_bounds__(256) void add_layernorm_kernel(
    const float* __restrict__ resid, const float* __restrict__ yv,
    const float* __restrict__ g, const float* __restrict__ beta,
    float* __restrict__ out, bf16* __restrict__ out_bf, int Dn)
{
  __shared__ float sh[8], sh2[8];
  const long long row = blockIdx.x;
  const float* r0 = resid + row * Dn;
  const float* y0 = yv + row * Dn;
  float* o0 = out + row * Dn;
  bf16* ob = out_bf ? out_bf + row * Dn : nullptr;
  const int tid = threadIdx.x, lane = tid & 31, wid = tid >> 5;

  float s = 0.f, s2 = 0.f;
  for (int i = tid; i < Dn; i += 256) {
    float v = r0[i] + y0[i];
    s += v; s2 += v * v;
  }
  s = wave_sum(s); s2 = wave_sum(s2);
  if (lane == 0) { sh[wid] = s; sh2[wid] = s2; }
  __syncthreads();
  float ts = 0.f, ts2 = 0.f;
  for (int i = 0; i < 8; ++i) { ts += sh[i]; ts2 += sh2[i]; }
  const float mu  = ts / Dn;
  const float var = ts2 / Dn - mu * mu;
  const float rs  = rsqrtf(var + cfg::EPS);
  for (int i = tid; i < Dn; i += 256) {
    float v = (r0[i] + y0[i] - mu) * rs * g[i] + beta[i];
    o0[i] = v;
    if (ob) ob[i] = f2bf(v);
  }
}

// ---------------------------------------------------------------------------
// Host-side GEMM dispatcher
// ---------------------------------------------------------------------------
template <typename CT>
static void launch_gemm(hipStream_t st, bool bt, bool relu,
    const bf16* A, int lda, long long sAb,
    const bf16* Bm, int ldb, long long sBb,
    const float* bias,
    CT* C, int ldc, long long sCb,
    int M, int N, int K, int batches)
{
  dim3 grid((N + 127) / 128, (M + 127) / 128, batches);
  dim3 blk(256);
  if (bt) {
    if (relu) gemm_bf16_kernel<true , true , CT><<<grid, blk, 0, st>>>(A, lda, sAb, Bm, ldb, sBb, bias, C, ldc, sCb, M, N, K);
    else      gemm_bf16_kernel<true , false, CT><<<grid, blk, 0, st>>>(A, lda, sAb, Bm, ldb, sBb, bias, C, ldc, sCb, M, N, K);
  } else {
    if (relu) gemm_bf16_kernel<false, true , CT><<<grid, blk, 0, st>>>(A, lda, sAb, Bm, ldb, sBb, bias, C, ldc, sCb, M, N, K);
    else      gemm_bf16_kernel<false, false, CT><<<grid, blk, 0, st>>>(A, lda, sAb, Bm, ldb, sBb, bias, C, ldc, sCb, M, N, K);
  }
}

extern "C" void kernel_launch(void* const* d_in, const int* in_sizes, int n_in,
                              void* d_out, int out_size, void* d_ws, size_t ws_size,
                              hipStream_t stream)
{
  using namespace cfg;
  (void)in_sizes; (void)n_in; (void)out_size; (void)ws_size;

  const float* tgt      = (const float*)d_in[0];
  const float* memory   = (const float*)d_in[1];
  const float* sa_in_w  = (const float*)d_in[2];
  const float* sa_in_b  = (const float*)d_in[3];
  const float* sa_out_w = (const float*)d_in[4];
  const float* sa_out_b = (const float*)d_in[5];
  const float* ca_in_w  = (const float*)d_in[6];
  const float* ca_in_b  = (const float*)d_in[7];
  const float* ca_out_w = (const float*)d_in[8];
  const float* ca_out_b = (const float*)d_in[9];
  const float* norm1_g  = (const float*)d_in[10];
  const float* norm1_b  = (const float*)d_in[11];
  const float* norm2_g  = (const float*)d_in[12];
  const float* norm2_b  = (const float*)d_in[13];
  const float* norm3_g  = (const float*)d_in[14];
  const float* norm3_b  = (const float*)d_in[15];
  const float* lin_w    = (const float*)d_in[16];
  const float* lin_b    = (const float*)d_in[17];
  const float* phi      = (const float*)d_in[18];
  const float* exp_w    = (const float*)d_in[19];
  const float* exp_b    = (const float*)d_in[20];

  const int rows = B * M;   // 4096 token rows
  typedef long long ll;

  char* base = (char*)d_ws;
  size_t off = 0;
  auto alloc_bf = [&](size_t elems) {
    bf16* p = (bf16*)(base + off);
    off += ((elems * 2 + 255) & ~(size_t)255);
    return p;
  };
  auto alloc_f = [&](size_t elems) {
    float* p = (float*)(base + off);
    off += ((elems * 4 + 255) & ~(size_t)255);
    return p;
  };

  // bf16 workspace
  bf16* tgt_bf  = alloc_bf((size_t)rows * D);
  bf16* mem_bf  = alloc_bf((size_t)B * S * D);
  bf16* wsaT    = alloc_bf((size_t)D * 3 * D);     // (D, 3D)
  bf16* wsoT    = alloc_bf((size_t)D * D);
  bf16* wcaT    = alloc_bf((size_t)D * 3 * D);
  bf16* wcoT    = alloc_bf((size_t)D * D);
  bf16* wlinT   = alloc_bf((size_t)D * FF);        // (D, FF)
  bf16* phi_bf  = alloc_bf((size_t)FF * NE);
  bf16* qkv_bf  = alloc_bf((size_t)rows * 3 * D);  // reused: qc (rows*D) + kvc (B*S*2D)
  bf16* probs_bf= alloc_bf((size_t)B * M * S);
  bf16* obuf_bf = alloc_bf((size_t)rows * D);
  bf16* x1_bf   = alloc_bf((size_t)rows * D);
  bf16* x2_bf   = alloc_bf((size_t)rows * D);
  bf16* hbuf_bf = alloc_bf((size_t)rows * FF);
  bf16* dispT_bf= alloc_bf((size_t)rows * NE);
  bf16* comb_bf = alloc_bf((size_t)rows * NE);
  bf16* slots_bf= alloc_bf((size_t)B * NE * FF);
  bf16* ybuf_bf = alloc_bf((size_t)B * NE * D);
  // f32 workspace
  float* scores = alloc_f((size_t)B * M * S);
  float* tmp    = alloc_f((size_t)rows * D);
  float* x1     = alloc_f((size_t)rows * D);       // reused as ff after LN2
  float* x2     = alloc_f((size_t)rows * D);
  float* logits = alloc_f((size_t)rows * NE);

  // ---------------- one-time bf16 conversions ----------------
  auto conv = [&](const float* s, bf16* d, size_t n) {
    ll n4 = (ll)(n / 4);
    ll b = (n4 + 255) / 256; if (b > 8192) b = 8192;
    convert_bf16_kernel<<<(int)b, 256, 0, stream>>>(s, d, n4);
  };
  auto convT = [&](const float* s, bf16* d, int N_, int K_) {
    dim3 g((K_ + 31) / 32, (N_ + 31) / 32);
    transpose_convert_kernel<<<g, 256, 0, stream>>>(s, d, N_, K_);
  };
  conv(tgt, tgt_bf, (size_t)rows * D);
  conv(memory, mem_bf, (size_t)B * S * D);
  conv(phi, phi_bf, (size_t)FF * NE);
  convT(sa_in_w,  wsaT, 3 * D, D);
  convT(sa_out_w, wsoT, D, D);
  convT(ca_in_w,  wcaT, 3 * D, D);
  convT(ca_out_w, wcoT, D, D);
  convT(lin_w,    wlinT, FF, D);

  // ============================ Self attention ============================
  launch_gemm<bf16>(stream, false, false,
      tgt_bf, D, 0, wsaT, 3 * D, 0, sa_in_b,
      qkv_bf, 3 * D, 0, rows, 3 * D, D, 1);

  for (int h = 0; h < H; ++h) {
    launch_gemm<float>(stream, true, false,
        qkv_bf + h * HD,     3 * D, (ll)M * 3 * D,
        qkv_bf + D + h * HD, 3 * D, (ll)M * 3 * D,
        nullptr,
        scores, S, (ll)M * S, M, S, HD, B);
    softmax_rows_kernel<<<B * M, 256, 0, stream>>>(scores, probs_bf, S, 0.125f);
    launch_gemm<bf16>(stream, false, false,
        probs_bf, S, (ll)M * S,
        qkv_bf + 2 * D + h * HD, 3 * D, (ll)M * 3 * D,
        nullptr,
        obuf_bf + h * HD, D, (ll)M * D, M, HD, S, B);
  }
  launch_gemm<float>(stream, false, false,
      obuf_bf, D, 0, wsoT, D, 0, sa_out_b,
      tmp, D, 0, rows, D, D, 1);
  add_layernorm_kernel<<<rows, 256, 0, stream>>>(tgt, tmp, norm1_g, norm1_b,
                                                 x1, x1_bf, D);

  // ============================ Cross attention ===========================
  bf16* qc  = qkv_bf;                          // (rows, D)
  bf16* kvc = qkv_bf + (size_t)rows * D;       // (B*S, 2D)
  launch_gemm<bf16>(stream, false, false,
      x1_bf, D, 0, wcaT, 3 * D, 0, ca_in_b,
      qc, D, 0, rows, D, D, 1);
  launch_gemm<bf16>(stream, false, false,
      mem_bf, D, 0, wcaT + D, 3 * D, 0, ca_in_b + D,
      kvc, 2 * D, 0, B * S, 2 * D, D, 1);

  for (int h = 0; h < H; ++h) {
    launch_gemm<float>(stream, true, false,
        qc + h * HD,  D,     (ll)M * D,
        kvc + h * HD, 2 * D, (ll)S * 2 * D,
        nullptr,
        scores, S, (ll)M * S, M, S, HD, B);
    softmax_rows_kernel<<<B * M, 256, 0, stream>>>(scores, probs_bf, S, 0.125f);
    launch_gemm<bf16>(stream, false, false,
        probs_bf, S, (ll)M * S,
        kvc + D + h * HD, 2 * D, (ll)S * 2 * D,
        nullptr,
        obuf_bf + h * HD, D, (ll)M * D, M, HD, S, B);
  }
  launch_gemm<float>(stream, false, false,
      obuf_bf, D, 0, wcoT, D, 0, ca_out_b,
      tmp, D, 0, rows, D, D, 1);
  add_layernorm_kernel<<<rows, 256, 0, stream>>>(x1, tmp, norm2_g, norm2_b,
                                                 x2, x2_bf, D);

  // ============================ FFN + SoftMoE =============================
  launch_gemm<bf16>(stream, false, true,
      x2_bf, D, 0, wlinT, FF, 0, lin_b,
      hbuf_bf, FF, 0, rows, FF, D, 1);

  launch_gemm<float>(stream, false, false,
      hbuf_bf, FF, 0, phi_bf, NE, 0, nullptr,
      logits, NE, 0, rows, NE, FF, 1);

  dispatch_softmax_kernel<<<B * NE, 256, 0, stream>>>(logits, dispT_bf, M);
  softmax_rows_kernel<<<rows, 256, 0, stream>>>(logits, comb_bf, NE, 1.0f);

  launch_gemm<bf16>(stream, false, false,
      dispT_bf, M, (ll)NE * M,
      hbuf_bf, FF, (ll)M * FF,
      nullptr,
      slots_bf, FF, (ll)NE * FF, NE, FF, M, B);

  {
    dim3 grid(D / 128, 1, NE);
    gemm_expert_kernel<<<grid, 256, 0, stream>>>(slots_bf, exp_w, exp_b,
                                                 ybuf_bf, B, D, FF);
  }

  float* ff = x1;   // x1 free after LN2
  launch_gemm<float>(stream, false, false,
      comb_bf, NE, (ll)M * NE,
      ybuf_bf, D, (ll)NE * D,
      nullptr,
      ff, D, (ll)M * D, M, D, NE, B);

  add_layernorm_kernel<<<rows, 256, 0, stream>>>(x2, ff, norm3_g, norm3_b,
                                                 (float*)d_out, nullptr, D);
}
